// MultiHeadAttn_72859825209646
// MI455X (gfx1250) — matrix operand order repaired and checked by reading, not compiled, not run on hardware
//
#include <hip/hip_runtime.h>

// ---------------------------------------------------------------------------
// CDNA5 (gfx1250, wave32) multi-head causal attention forward.
// Matmuls via v_wmma_f32_16x16x32_bf16; global->LDS staging via the async
// DMA path (GLOBAL_LOAD_ASYNC_TO_LDS_B128 / ASYNCcnt) with double buffering
// when the toolchain exposes the builtins.
// ---------------------------------------------------------------------------

typedef __bf16 bf16;
typedef __attribute__((ext_vector_type(16))) bf16  v16bf;
typedef __attribute__((ext_vector_type(8)))  bf16  v8bf;
typedef __attribute__((ext_vector_type(8)))  float v8f;

#define D_MODEL 2048
#define N_HEADS 16
#define D_HEAD  128
#define BATCH   2
#define SEQ     2048
#define BS      (BATCH * SEQ)                 // 4096 token rows
#define QK_SCALE 0.08838834764831845f        // 1/sqrt(128)
#define NEG_INF (-__builtin_inff())

#if __has_builtin(__builtin_amdgcn_global_load_async_to_lds_b128) && \
    __has_builtin(__builtin_amdgcn_s_wait_asynccnt)
#define USE_ASYNC 1
#pragma message("gfx1250: using GLOBAL_LOAD_ASYNC_TO_LDS_B128 double-buffer path")
#else
#define USE_ASYNC 0
#pragma message("gfx1250: async-to-LDS builtins unavailable; sync LDS staging")
#endif

#if USE_ASYNC
// Builtin signature (from hipcc diagnostic): parameters are pointers to
// 'int __attribute__((vector_size(16)))' (generic int4*).
typedef int v4i_vs __attribute__((vector_size(16)));
// One 16-byte async DMA transfer per lane: global -> LDS, no VGPR round-trip.
static __device__ __forceinline__ void async_b128(const void* g, void* l) {
  __builtin_amdgcn_global_load_async_to_lds_b128(
      (v4i_vs*)(uintptr_t)g, (v4i_vs*)(uintptr_t)l, 0, 0);
}
static __device__ __forceinline__ void wait_async0() {
  __builtin_amdgcn_s_wait_asynccnt(0);
}
#endif

// Stage 16 bf16 (32B) from global to LDS; async when available.
static __device__ __forceinline__ void stage32B(const bf16* g, bf16* l) {
#if USE_ASYNC
  async_b128(g, l);
  async_b128(g + 8, l + 8);
#else
  *(v8bf*)l       = *(const v8bf*)g;
  *(v8bf*)(l + 8) = *(const v8bf*)(g + 8);
#endif
}

static __device__ __forceinline__ v8f wmma_bf16(v16bf a, v16bf b, v8f c) {
  return __builtin_amdgcn_wmma_f32_16x16x32_bf16(false, a, false, b, (short)0, c,
                                                 false, false);
}

// Assemble a 16-element bf16 fragment from two 16-byte chunks.
static __device__ __forceinline__ v16bf ld_frag(const bf16* p0, const bf16* p1) {
  union { v16bf v; v8bf h[2]; } u;
  u.h[0] = *(const v8bf*)p0;
  u.h[1] = *(const v8bf*)p1;
  return u.v;
}

// ---------------------------------------------------------------------------
// Stage 0a: fp32 -> bf16 (for X)
// ---------------------------------------------------------------------------
__global__ void cvt_f32_bf16(const float* __restrict__ src,
                             bf16* __restrict__ dst, int n) {
  int i = (blockIdx.x * blockDim.x + threadIdx.x) * 4;
  if (i < n) {
    float4 v = *(const float4*)(src + i);
    dst[i + 0] = (bf16)v.x;
    dst[i + 1] = (bf16)v.y;
    dst[i + 2] = (bf16)v.z;
    dst[i + 3] = (bf16)v.w;
  }
}

// ---------------------------------------------------------------------------
// Stage 0b: transpose-convert W[k][n] f32  ->  Wt[n][k] bf16  (2048x2048)
// ---------------------------------------------------------------------------
__global__ void transpose_f32_bf16(const float* __restrict__ src,
                                   bf16* __restrict__ dst) {
  __shared__ float tile[32][33];
  int x = blockIdx.x * 32 + threadIdx.x;
  int y = blockIdx.y * 32 + threadIdx.y;
#pragma unroll
  for (int j = 0; j < 32; j += 8)
    tile[threadIdx.y + j][threadIdx.x] = src[(size_t)(y + j) * D_MODEL + x];
  __syncthreads();
  x = blockIdx.y * 32 + threadIdx.x;
  y = blockIdx.x * 32 + threadIdx.y;
#pragma unroll
  for (int j = 0; j < 32; j += 8)
    dst[(size_t)(y + j) * D_MODEL + x] = (bf16)tile[threadIdx.x][threadIdx.y + j];
}

// ---------------------------------------------------------------------------
// WMMA GEMM:  C[4096 x 2048] = A[4096 x 2048] @ Bt[2048 x 2048]^T + bias
// Workgroup tile 128x128, 8 waves (2x4), each wave 64x32 (4x2 WMMA tiles).
// Double-buffered LDS; async DMA fill of tile kc+1 overlaps WMMA on tile kc.
// MODE 0: Q   -> bf16 [b,h,s,d], scaled by 1/sqrt(d_head)
// MODE 1: K   -> bf16 [b,h,s,d]
// MODE 2: V   -> bf16 [b,h,d,s]  (transposed for PV B-fragments)
// MODE 3: out -> f32  [b,s,d_model]
// ---------------------------------------------------------------------------
template <int MODE>
__global__ __launch_bounds__(256) void gemm_bf16(const bf16* __restrict__ A,
                                                 const bf16* __restrict__ Bt,
                                                 const float* __restrict__ bias,
                                                 void* __restrict__ out) {
  const int K = D_MODEL;
  const int LDP = 40;  // padded pitch (bf16 units): 80B rows, spreads LDS banks
  __shared__ __attribute__((aligned(16))) bf16 Asm[2][128 * LDP];
  __shared__ __attribute__((aligned(16))) bf16 Bsm[2][128 * LDP];

  const int tid = threadIdx.x;
  const int lane = tid & 31, wave = tid >> 5;
  const int wm = wave >> 2, wn = wave & 3;
  const int half = lane >> 4, col = lane & 15;
  const int Mb = blockIdx.y * 128, Nb = blockIdx.x * 128;

  const v8f vz = {0.f, 0.f, 0.f, 0.f, 0.f, 0.f, 0.f, 0.f};
  v8f acc[4][2];
#pragma unroll
  for (int i = 0; i < 4; ++i)
#pragma unroll
    for (int j = 0; j < 2; ++j) acc[i][j] = vz;

  const int lrow = tid >> 1, lseg = tid & 1;  // 2 threads/row, 32B each
  const bf16* gA = A + (size_t)(Mb + lrow) * K + lseg * 16;
  const bf16* gB = Bt + (size_t)(Nb + lrow) * K + lseg * 16;
  const int lofs = lrow * LDP + lseg * 16;

  auto issue = [&](int buf, int k0) {
    stage32B(gA + k0, &Asm[buf][lofs]);
    stage32B(gB + k0, &Bsm[buf][lofs]);
  };

  issue(0, 0);
  const int NK = K / 32;  // 64 k-chunks
  for (int kc = 0; kc < NK; ++kc) {
    const int cur = kc & 1;
#if USE_ASYNC
    wait_async0();  // own DMA writes for buffer `cur` have landed
#endif
    __syncthreads();  // everyone's writes landed; no one reads buf cur^1
    if (kc + 1 < NK) issue(cur ^ 1, (kc + 1) * 32);

    v16bf af[4];
#pragma unroll
    for (int i = 0; i < 4; ++i) {
      // A frag: lane holds row m=col, k = (e>>3)*16 + half*8 + (e&7)
      const bf16* ap = &Asm[cur][(wm * 64 + i * 16 + col) * LDP + half * 8];
      af[i] = ld_frag(ap, ap + 16);
    }
#pragma unroll
    for (int j = 0; j < 2; ++j) {
      // B frag: lane holds col n=col, k = (e>>3)*16 + half*8 + (e&7)
      const bf16* bp = &Bsm[cur][(wn * 32 + j * 16 + col) * LDP + half * 8];
      v16bf bfr = ld_frag(bp, bp + 16);
#pragma unroll
      for (int i = 0; i < 4; ++i) acc[i][j] = wmma_bf16(af[i], bfr, acc[i][j]);
    }
  }

  // Epilogue: C layout rows r+8*half across VGPRs, cols = lane&15.
#pragma unroll
  for (int j = 0; j < 2; ++j) {
    const int gn = Nb + wn * 32 + j * 16 + col;
    const float bv = bias[gn];
#pragma unroll
    for (int i = 0; i < 4; ++i) {
#pragma unroll
      for (int r = 0; r < 8; ++r) {
        const int gm = Mb + wm * 64 + i * 16 + r + 8 * half;
        float v = acc[i][j][r] + bv;
        if constexpr (MODE == 3) {
          ((float*)out)[(size_t)gm * D_MODEL + gn] = v;
        } else {
          const int b = gm >> 11, s = gm & 2047;
          const int h = gn >> 7, d = gn & 127;
          if constexpr (MODE == 2) {
            ((bf16*)out)[(((size_t)(b * N_HEADS + h)) * D_HEAD + d) * SEQ + s] =
                (bf16)v;
          } else {
            if constexpr (MODE == 0) v *= QK_SCALE;
            ((bf16*)out)[(((size_t)(b * N_HEADS + h)) * SEQ + s) * D_HEAD + d] =
                (bf16)v;
          }
        }
      }
    }
  }
}

// ---------------------------------------------------------------------------
// Flash-style causal attention. Grid (16 q-blocks, 32 bh), 256 threads.
// Each wave owns 16 q rows; workgroup shares double-buffered K/V tiles
// (32 kv rows) staged via async DMA.
// ---------------------------------------------------------------------------
__global__ __launch_bounds__(256) void attn_fwd(const bf16* __restrict__ Q,
                                                const bf16* __restrict__ Kt,
                                                const bf16* __restrict__ Vt,
                                                bf16* __restrict__ O) {
  const int LDK = 136, LDV = 40, LDP = 40;  // padded pitches (bf16 units)
  __shared__ __attribute__((aligned(16))) bf16 Ksm[2][32 * 136];
  __shared__ __attribute__((aligned(16))) bf16 Vsm[2][128 * 40];
  __shared__ __attribute__((aligned(16))) bf16 Psm[8 * 16 * 40];

  const int tid = threadIdx.x, lane = tid & 31, w = tid >> 5;
  const int half = lane >> 4, col = lane & 15;
  const int bh = blockIdx.y;
  const int qbase = blockIdx.x * 128;
  const int qlo = qbase + w * 16;

  const bf16* Qb = Q + (size_t)bh * SEQ * D_HEAD;
  const bf16* Kb = Kt + (size_t)bh * SEQ * D_HEAD;
  const bf16* Vb = Vt + (size_t)bh * D_HEAD * SEQ;

  // Preload this wave's Q tile (16 x 128) as four 16x32 A-fragments.
  v16bf qf[4];
  {
    const bf16* qr = Qb + (size_t)(qlo + col) * D_HEAD;
#pragma unroll
    for (int i = 0; i < 4; ++i) {
      const bf16* p = qr + i * 32 + half * 8;
      qf[i] = ld_frag(p, p + 16);
    }
  }

  const v8f vz = {0.f, 0.f, 0.f, 0.f, 0.f, 0.f, 0.f, 0.f};
  v8f acc[8];  // O accumulator: 8 d-chunks of 16
#pragma unroll
  for (int j = 0; j < 8; ++j) acc[j] = vz;
  float mrow[8], lrow[8];
#pragma unroll
  for (int r = 0; r < 8; ++r) { mrow[r] = NEG_INF; lrow[r] = 0.f; }

  // Cooperative staging addresses (per thread).
  const int kr = tid >> 3, ks = tid & 7;   // K: 32 rows x 8 segs of 16 elems
  const int vd = tid >> 1, vs = tid & 1;   // V: 128 rows x 2 segs of 16 elems
  auto issueKV = [&](int buf, int kv0) {
    stage32B(Kb + (size_t)(kv0 + kr) * D_HEAD + ks * 16,
             &Ksm[buf][kr * LDK + ks * 16]);
    stage32B(Vb + (size_t)vd * SEQ + kv0 + vs * 16,
             &Vsm[buf][vd * LDV + vs * 16]);
  };

  const int nt = qbase / 32 + 4;  // kv tiles (32 rows) covering causal extent
  issueKV(0, 0);
  for (int t = 0; t < nt; ++t) {
    const int kv0 = t * 32;
    const int cur = t & 1;
#if USE_ASYNC
    wait_async0();
#endif
    __syncthreads();
    if (t + 1 < nt) issueKV(cur ^ 1, (t + 1) * 32);
    if (kv0 > qlo + 15) continue;  // fully above diagonal for this wave

    // S tile (16 q x 32 kv) = Q @ K^T : two 16x16 C tiles, 4 WMMAs each.
    v8f s0 = vz, s1 = vz;
#pragma unroll
    for (int i = 0; i < 4; ++i) {
      const bf16* bp0 = &Ksm[cur][col * LDK + i * 32 + half * 8];
      s0 = wmma_bf16(qf[i], ld_frag(bp0, bp0 + 16), s0);
      const bf16* bp1 = &Ksm[cur][(16 + col) * LDK + i * 32 + half * 8];
      s1 = wmma_bf16(qf[i], ld_frag(bp1, bp1 + 16), s1);
    }

    // Online softmax (state replicated across the 16 lanes of each half).
    float p0[8], p1[8], alpha[8];
#pragma unroll
    for (int r = 0; r < 8; ++r) {
      const int q = qlo + r + 8 * half;
      float v0 = (kv0 + col <= q) ? s0[r] : NEG_INF;
      float v1 = (kv0 + 16 + col <= q) ? s1[r] : NEG_INF;
      float vm = fmaxf(v0, v1);
      vm = fmaxf(vm, __shfl_xor(vm, 1, 32));
      vm = fmaxf(vm, __shfl_xor(vm, 2, 32));
      vm = fmaxf(vm, __shfl_xor(vm, 4, 32));
      vm = fmaxf(vm, __shfl_xor(vm, 8, 32));
      const float mnew = fmaxf(mrow[r], vm);
      const float a = __expf(mrow[r] - mnew);
      mrow[r] = mnew;
      alpha[r] = a;
      p0[r] = __expf(v0 - mnew);
      p1[r] = __expf(v1 - mnew);
      float ts = p0[r] + p1[r];
      ts += __shfl_xor(ts, 1, 32);
      ts += __shfl_xor(ts, 2, 32);
      ts += __shfl_xor(ts, 4, 32);
      ts += __shfl_xor(ts, 8, 32);
      lrow[r] = lrow[r] * a + ts;
    }
#pragma unroll
    for (int j = 0; j < 8; ++j)
#pragma unroll
      for (int r = 0; r < 8; ++r) acc[j][r] *= alpha[r];

    // Stage P (C layout) -> LDS -> reload as one 16x32 A-fragment.
    bf16* P = &Psm[w * 16 * LDP];
#pragma unroll
    for (int r = 0; r < 8; ++r) {
      const int m = r + 8 * half;
      P[m * LDP + col]      = (bf16)p0[r];
      P[m * LDP + 16 + col] = (bf16)p1[r];
    }
    // Same-wave DS ordering: drain DScnt before reading staged P.
    asm volatile("s_wait_dscnt 0x0" ::: "memory");
    const bf16* ap = &P[col * LDP + half * 8];
    const v16bf pa = ld_frag(ap, ap + 16);
#pragma unroll
    for (int j = 0; j < 8; ++j) {
      const bf16* bp = &Vsm[cur][(j * 16 + col) * LDV + half * 8];
      acc[j] = wmma_bf16(pa, ld_frag(bp, bp + 16), acc[j]);
    }
  }

  // Normalize and write attn output as bf16 [b, s, h*128+d].
  const int b = bh >> 4, h = bh & 15;
#pragma unroll
  for (int r = 0; r < 8; ++r) {
    const float inv = 1.0f / lrow[r];
    const int q = qlo + r + 8 * half;
    bf16* orow = O + ((size_t)(b * SEQ + q)) * D_MODEL + h * D_HEAD;
#pragma unroll
    for (int j = 0; j < 8; ++j) orow[j * 16 + col] = (bf16)(acc[j][r] * inv);
  }
}

// ---------------------------------------------------------------------------
extern "C" void kernel_launch(void* const* d_in, const int* in_sizes, int n_in,
                              void* d_out, int out_size, void* d_ws,
                              size_t ws_size, hipStream_t stream) {
  const float* X  = (const float*)d_in[0];
  const float* Wq = (const float*)d_in[1];
  const float* bq = (const float*)d_in[2];
  const float* Wk = (const float*)d_in[3];
  const float* bk = (const float*)d_in[4];
  const float* Wv = (const float*)d_in[5];
  const float* bv = (const float*)d_in[6];
  const float* Wo = (const float*)d_in[7];
  const float* bo = (const float*)d_in[8];

  char* ws = (char*)d_ws;
  size_t off = 0;
  auto alloc = [&](size_t bytes) {
    void* p = ws + off;
    off += (bytes + 255) & ~(size_t)255;
    return p;
  };
  bf16* Xb  = (bf16*)alloc((size_t)BS * D_MODEL * 2);       // 16 MB
  bf16* Wqt = (bf16*)alloc((size_t)D_MODEL * D_MODEL * 2);  //  8 MB
  bf16* Wkt = (bf16*)alloc((size_t)D_MODEL * D_MODEL * 2);
  bf16* Wvt = (bf16*)alloc((size_t)D_MODEL * D_MODEL * 2);
  bf16* Wot = (bf16*)alloc((size_t)D_MODEL * D_MODEL * 2);
  bf16* Qw  = (bf16*)alloc((size_t)BS * D_MODEL * 2);       // [b,h,s,d]
  bf16* Kw  = (bf16*)alloc((size_t)BS * D_MODEL * 2);       // [b,h,s,d]
  bf16* Vtw = (bf16*)alloc((size_t)BS * D_MODEL * 2);       // [b,h,d,s]
  bf16* Ab  = (bf16*)alloc((size_t)BS * D_MODEL * 2);       // [b,s,dm]
  (void)ws_size; (void)in_sizes; (void)n_in; (void)out_size;

  // Stage 0: precision conversion / weight transposes (bandwidth-trivial).
  cvt_f32_bf16<<<(BS * D_MODEL / 4 + 255) / 256, 256, 0, stream>>>(
      X, Xb, BS * D_MODEL);
  dim3 tb(32, 8), tg(64, 64);
  transpose_f32_bf16<<<tg, tb, 0, stream>>>(Wq, Wqt);
  transpose_f32_bf16<<<tg, tb, 0, stream>>>(Wk, Wkt);
  transpose_f32_bf16<<<tg, tb, 0, stream>>>(Wv, Wvt);
  transpose_f32_bf16<<<tg, tb, 0, stream>>>(Wo, Wot);

  // Stage 1: QKV projections (WMMA).
  dim3 gg(16, 32);  // (N/128, M/128)
  gemm_bf16<0><<<gg, 256, 0, stream>>>(Xb, Wqt, bq, Qw);
  gemm_bf16<1><<<gg, 256, 0, stream>>>(Xb, Wkt, bk, Kw);
  gemm_bf16<2><<<gg, 256, 0, stream>>>(Xb, Wvt, bv, Vtw);

  // Stage 2: causal flash attention (WMMA for QK^T and PV).
  dim3 ag(16, 32);  // (q-blocks of 128, b*h)
  attn_fwd<<<ag, 256, 0, stream>>>(Qw, Kw, Vtw, Ab);

  // Stage 3: output projection -> f32 d_out.
  gemm_bf16<3><<<gg, 256, 0, stream>>>(Ab, Wot, bo, d_out);
}
